// SelfAttention_87187836108851
// MI455X (gfx1250) — matrix operand order repaired and checked by reading, not compiled, not run on hardware
//
#include <hip/hip_runtime.h>
#include <hip/hip_bf16.h>

#define B_ 16
#define S_ 2048
#define H_ 512
#define A_ 256
#define NCH 8
#define CHLEN (S_ / NCH)   // 256

typedef __attribute__((ext_vector_type(16))) __bf16 v16bf;
typedef __attribute__((ext_vector_type(8)))  __bf16 v8bf;
typedef __attribute__((ext_vector_type(8)))  float  v8f;

// ---------------------------------------------------------------------------
// Pre-split: f32 -> bf16 hi (truncate) + bf16 lo (residual, truncated).
// hi + lo reproduces f to ~2^-16 relative: hi = f&0xffff0000 (error <= 2^-8),
// residual r = f - hi is exact in f32, lo = trunc16(r) (error 2^-8 of r).
// 8 elements per thread, packed uint4 stores.
// ---------------------------------------------------------------------------
__global__ __launch_bounds__(256) void split_kernel(
    const float* __restrict__ src,
    unsigned short* __restrict__ hi, unsigned short* __restrict__ lo)
{
    const size_t idx = ((size_t)blockIdx.x * 256 + threadIdx.x) * 8;
    const float4* s4 = (const float4*)(src + idx);
    float4 f0 = s4[0], f1 = s4[1];
    float f[8] = {f0.x, f0.y, f0.z, f0.w, f1.x, f1.y, f1.z, f1.w};
    unsigned hp[4], lp[4];
#pragma unroll
    for (int i = 0; i < 4; ++i) {
        unsigned u0 = __builtin_bit_cast(unsigned, f[2 * i]);
        unsigned u1 = __builtin_bit_cast(unsigned, f[2 * i + 1]);
        float r0 = f[2 * i]     - __builtin_bit_cast(float, u0 & 0xffff0000u);
        float r1 = f[2 * i + 1] - __builtin_bit_cast(float, u1 & 0xffff0000u);
        unsigned l0 = __builtin_bit_cast(unsigned, r0) >> 16;
        unsigned l1 = __builtin_bit_cast(unsigned, r1) >> 16;
        hp[i] = (u0 >> 16) | (u1 & 0xffff0000u);
        lp[i] = l0 | (l1 << 16);
    }
    uint4 hv = make_uint4(hp[0], hp[1], hp[2], hp[3]);
    uint4 lv = make_uint4(lp[0], lp[1], lp[2], lp[3]);
    *(uint4*)(hi + idx) = hv;
    *(uint4*)(lo + idx) = lv;
}

__device__ __forceinline__ v16bf load_a_tile(const unsigned short* base) {
    // elements 0..7 at +0, elements 8..15 at +8 (contiguous K)
    v8bf p0 = *(const v8bf*)(base);
    v8bf p1 = *(const v8bf*)(base + 8);
    return __builtin_shufflevector(p0, p1, 0, 1, 2, 3, 4, 5, 6, 7,
                                           8, 9, 10, 11, 12, 13, 14, 15);
}

// ---------------------------------------------------------------------------
// Kernel 1: e[row] = sum_a q[a] * tanh( sum_h x[row,h] * W_a[a,h] )
// GEMM [B*S,H] x [H,A] via V_WMMA_F32_16X16X32_BF16, bf16 hi/lo split
// (hi*hi + hi*lo + lo*hi = 3 WMMAs per 16x16x32 tile, ~fp32 accuracy).
// Block = 128 threads (4 waves), covers 32 rows (2 A-tiles) x 256 cols.
// Each wave owns 4 of the 16 N-tiles; B-tile load feeds 6 WMMAs.
// ---------------------------------------------------------------------------
__global__ __launch_bounds__(128) void score_gemm_kernel(
    const unsigned short* __restrict__ xh, const unsigned short* __restrict__ xl,
    const unsigned short* __restrict__ wh, const unsigned short* __restrict__ wl,
    const float* __restrict__ q, float* __restrict__ e_out)
{
    __shared__ float e_sh[32];
    const int tid = threadIdx.x;
    if (tid < 32) e_sh[tid] = 0.0f;
    __syncthreads();

    const int lane = tid & 31;
    const int wave = tid >> 5;
    const int m    = lane & 15;   // A: row within tile; B: col N
    const int kg   = lane >> 4;   // K sub-group select (contiguous K order, k = 16*kg + s)
    const int row0 = blockIdx.x * 32;

    v8f acc[2][4];
#pragma unroll
    for (int rt = 0; rt < 2; ++rt)
#pragma unroll
        for (int j = 0; j < 4; ++j)
#pragma unroll
            for (int r = 0; r < 8; ++r) acc[rt][j][r] = 0.0f;

    // A rows for the two row-tiles this block covers
    const unsigned short* xh0 = xh + (size_t)(row0 + m) * H_;
    const unsigned short* xl0 = xl + (size_t)(row0 + m) * H_;
    const unsigned short* xh1 = xh0 + (size_t)16 * H_;
    const unsigned short* xl1 = xl0 + (size_t)16 * H_;

    for (int k0 = 0; k0 < H_; k0 += 32) {
        // A tiles: lanes 0-15 K=0..15; lanes 16-31 K=16..31
        const int ka = k0 + kg * 16;
        v16bf a_hi0 = load_a_tile(xh0 + ka);
        v16bf a_lo0 = load_a_tile(xl0 + ka);
        v16bf a_hi1 = load_a_tile(xh1 + ka);
        v16bf a_lo1 = load_a_tile(xl1 + ka);

#pragma unroll
        for (int j = 0; j < 4; ++j) {
            const int a0 = (wave * 4 + j) * 16;
            // B tile: lanes 0-15 col N=m K=0..15; lanes 16-31 K=16..31
            const size_t boff = (size_t)(a0 + m) * H_ + k0 + kg * 16;
            v16bf b_hi = *(const v16bf*)(wh + boff);
            v16bf b_lo = *(const v16bf*)(wl + boff);

            acc[0][j] = __builtin_amdgcn_wmma_f32_16x16x32_bf16(
                false, a_hi0, false, b_hi, (short)0, acc[0][j], false, false);
            acc[0][j] = __builtin_amdgcn_wmma_f32_16x16x32_bf16(
                false, a_hi0, false, b_lo, (short)0, acc[0][j], false, false);
            acc[0][j] = __builtin_amdgcn_wmma_f32_16x16x32_bf16(
                false, a_lo0, false, b_hi, (short)0, acc[0][j], false, false);

            acc[1][j] = __builtin_amdgcn_wmma_f32_16x16x32_bf16(
                false, a_hi1, false, b_hi, (short)0, acc[1][j], false, false);
            acc[1][j] = __builtin_amdgcn_wmma_f32_16x16x32_bf16(
                false, a_hi1, false, b_lo, (short)0, acc[1][j], false, false);
            acc[1][j] = __builtin_amdgcn_wmma_f32_16x16x32_bf16(
                false, a_lo1, false, b_hi, (short)0, acc[1][j], false, false);
        }
    }

    // Epilogue: C layout -> VGPR r holds M = r + 8*kg, N = m.
#pragma unroll
    for (int rt = 0; rt < 2; ++rt) {
#pragma unroll
        for (int r = 0; r < 8; ++r) {
            float s = 0.0f;
#pragma unroll
            for (int j = 0; j < 4; ++j) {
                const int a0 = (wave * 4 + j) * 16;
                s += q[a0 + m] * tanhf(acc[rt][j][r]);
            }
#pragma unroll
            for (int off = 1; off < 16; off <<= 1)
                s += __shfl_xor(s, off, 16);
            if (m == 0) atomicAdd(&e_sh[rt * 16 + kg * 8 + r], s);
        }
    }
    __syncthreads();
    if (tid < 32) e_out[row0 + tid] = e_sh[tid];
}

// ---------------------------------------------------------------------------
// Kernel 2: per batch b -> M = max_t e, w[t] = exp(e-M),
//           invZ[s] = 1 / prefix_sum_{t<=s} w[t].
// ---------------------------------------------------------------------------
__global__ __launch_bounds__(256) void softmax_prefix_kernel(
    const float* __restrict__ e, float* __restrict__ w, float* __restrict__ invZ)
{
    const int b = blockIdx.x;
    const int tid = threadIdx.x;
    __shared__ float red[256];

    const float* eb = e + b * S_;
    float loc[8];
    float mx = -3.4e38f;
#pragma unroll
    for (int i = 0; i < 8; ++i) { loc[i] = eb[tid * 8 + i]; mx = fmaxf(mx, loc[i]); }
    red[tid] = mx; __syncthreads();
    for (int off = 128; off > 0; off >>= 1) {
        if (tid < off) red[tid] = fmaxf(red[tid], red[tid + off]);
        __syncthreads();
    }
    const float M = red[0];
    __syncthreads();

    float run = 0.0f;
#pragma unroll
    for (int i = 0; i < 8; ++i) { loc[i] = expf(loc[i] - M); run += loc[i]; }
#pragma unroll
    for (int i = 0; i < 8; ++i) w[b * S_ + tid * 8 + i] = loc[i];

    red[tid] = run; __syncthreads();
    for (int off = 1; off < 256; off <<= 1) {
        float v = red[tid];
        float add = (tid >= off) ? red[tid - off] : 0.0f;
        __syncthreads();
        red[tid] = v + add;
        __syncthreads();
    }
    float acc = (tid > 0) ? red[tid - 1] : 0.0f;
#pragma unroll
    for (int i = 0; i < 8; ++i) {
        acc += loc[i];
        invZ[b * S_ + tid * 8 + i] = 1.0f / acc;
    }
}

// ---------------------------------------------------------------------------
// Kernel 3: d[b,s,t] = (t<=s) ? w[b,t]*invZ[b,s] : 0    (268 MB, BW-bound)
// ---------------------------------------------------------------------------
__global__ __launch_bounds__(256) void write_d_kernel(
    const float* __restrict__ w, const float* __restrict__ invZ,
    float* __restrict__ d)
{
    const int s = blockIdx.x;
    const int b = blockIdx.y;
    const float iz = invZ[b * S_ + s];
    const float4* w4 = (const float4*)(w + (size_t)b * S_);
    float4* drow = (float4*)(d + ((size_t)b * S_ + s) * S_);
    for (int v = threadIdx.x; v < S_ / 4; v += 256) {
        const int t = v * 4;
        float4 ww = w4[v];
        float4 o;
        o.x = (t + 0 <= s) ? ww.x * iz : 0.0f;
        o.y = (t + 1 <= s) ? ww.y * iz : 0.0f;
        o.z = (t + 2 <= s) ? ww.z * iz : 0.0f;
        o.w = (t + 3 <= s) ? ww.w * iz : 0.0f;
        drow[v] = o;
    }
}

// ---------------------------------------------------------------------------
// Kernel 4a: per-chunk partials  P[b,ch,h] = sum_{t in chunk} w[b,t]*x[b,t,h]
// ---------------------------------------------------------------------------
__global__ __launch_bounds__(256) void partial_kernel(
    const float* __restrict__ x, const float* __restrict__ w,
    float* __restrict__ partial)
{
    const int ch = blockIdx.x, b = blockIdx.y, tid = threadIdx.x;
    const int h0 = tid, h1 = tid + 256;
    const float* xb = x + (size_t)b * S_ * H_;
    const float* wb = w + (size_t)b * S_;
    float a0 = 0.0f, a1 = 0.0f;
    const int tbeg = ch * CHLEN;
    for (int t = tbeg; t < tbeg + CHLEN; ++t) {
        const float wt = wb[t];
        a0 += wt * xb[(size_t)t * H_ + h0];
        a1 += wt * xb[(size_t)t * H_ + h1];
    }
    partial[((size_t)b * NCH + ch) * H_ + h0] = a0;
    partial[((size_t)b * NCH + ch) * H_ + h1] = a1;
}

// ---------------------------------------------------------------------------
// Kernel 4b: a[b,s,h] = invZ[b,s] * (sum_{chunks<ch} P + running sum in chunk)
// ---------------------------------------------------------------------------
__global__ __launch_bounds__(256) void output_a_kernel(
    const float* __restrict__ x, const float* __restrict__ w,
    const float* __restrict__ invZ, const float* __restrict__ partial,
    float* __restrict__ aout)
{
    const int ch = blockIdx.x, b = blockIdx.y, tid = threadIdx.x;
    const int h0 = tid, h1 = tid + 256;
    float a0 = 0.0f, a1 = 0.0f;
    for (int c = 0; c < ch; ++c) {
        a0 += partial[((size_t)b * NCH + c) * H_ + h0];
        a1 += partial[((size_t)b * NCH + c) * H_ + h1];
    }
    const float* xb  = x    + (size_t)b * S_ * H_;
    const float* wb  = w    + (size_t)b * S_;
    const float* izb = invZ + (size_t)b * S_;
    float* ab        = aout + (size_t)b * S_ * H_;
    const int tbeg = ch * CHLEN;
    for (int t = tbeg; t < tbeg + CHLEN; ++t) {
        const float wt = wb[t];
        a0 += wt * xb[(size_t)t * H_ + h0];
        a1 += wt * xb[(size_t)t * H_ + h1];
        const float iz = izb[t];
        ab[(size_t)t * H_ + h0] = a0 * iz;
        ab[(size_t)t * H_ + h1] = a1 * iz;
    }
}

extern "C" void kernel_launch(void* const* d_in, const int* in_sizes, int n_in,
                              void* d_out, int out_size, void* d_ws, size_t ws_size,
                              hipStream_t stream) {
    (void)in_sizes; (void)n_in; (void)out_size; (void)ws_size;
    const float* x  = (const float*)d_in[0];   // [B,S,H]
    const float* Wa = (const float*)d_in[1];   // [A,H]
    const float* q  = (const float*)d_in[2];   // [1,A]

    float* a_out = (float*)d_out;                          // [B,S,1,H]
    float* d_mat = (float*)d_out + (size_t)B_ * S_ * H_;   // [B,S,1,S]

    // workspace layout
    float* e       = (float*)d_ws;               // B*S f32
    float* w       = e    + (size_t)B_ * S_;     // B*S f32
    float* invZ    = w    + (size_t)B_ * S_;     // B*S f32
    float* partial = invZ + (size_t)B_ * S_;     // B*NCH*H f32
    unsigned short* xh = (unsigned short*)(partial + (size_t)B_ * NCH * H_); // B*S*H bf16
    unsigned short* xl = xh + (size_t)B_ * S_ * H_;                          // B*S*H bf16
    unsigned short* wh = xl + (size_t)B_ * S_ * H_;                          // A*H bf16
    unsigned short* wl = wh + (size_t)A_ * H_;                               // A*H bf16

    // 0) one-time bf16 hi/lo pre-split of x and W_a (bandwidth-bound)
    split_kernel<<<(B_ * S_ * H_) / (256 * 8), 256, 0, stream>>>(x, xh, xl);
    split_kernel<<<(A_ * H_) / (256 * 8), 256, 0, stream>>>(Wa, wh, wl);
    // 1) e[b,t] via pure-WMMA bf16-split GEMM
    score_gemm_kernel<<<(B_ * S_) / 32, 128, 0, stream>>>(xh, xl, wh, wl, q, e);
    // 2) prefix softmax terms
    softmax_prefix_kernel<<<B_, 256, 0, stream>>>(e, w, invZ);
    // 3) d matrix (pure bandwidth)
    write_d_kernel<<<dim3(S_, B_), 256, 0, stream>>>(w, invZ, d_mat);
    // 4) a via chunked prefix scan
    partial_kernel<<<dim3(NCH, B_), 256, 0, stream>>>(x, w, partial);
    output_a_kernel<<<dim3(NCH, B_), 256, 0, stream>>>(x, w, invZ, partial, a_out);
}
